// GDNN_22136261443720
// MI455X (gfx1250) — compile-verified, matrix operand order checked by reading, not run on hardware
//
#include <hip/hip_runtime.h>
#include <hip/hip_bf16.h>
#include <math.h>

// ---------------------------------------------------------------------------
// Equivariant 3-layer MLP for MI455X (gfx1250, wave32, WMMA).
//   Layer GEMMs: C(512x4096) = A(512xK) * W(4096xK)^T,  K = 1024 / 5120 / 9216
//   W materialized per-layer in f16 from (w, pattern) gathers; stays L2-resident
//   (192 MB L2 >= 75 MB max layer W). Activation relu(z)-z/2 == |z|/2 fused
//   into BN reduce/apply. GEMM: 64x256 block tile, 8 waves of 32x64 (2x4 WMMA
//   16x16x32 f16->f32), double-buffered LDS with register-staged global loads.
// Workspace (bytes, ~103 MB): W 75,497,472 | Aa 9,437,184 | Ab 9,437,184 |
//   O2 8,388,608 | stats 512
// ---------------------------------------------------------------------------

typedef __attribute__((ext_vector_type(16))) _Float16 v16h;
typedef __attribute__((ext_vector_type(8)))  _Float16 v8h;
typedef __attribute__((ext_vector_type(8)))  float    v8f;

union Frag16 { v16h v; v8h h[2]; };

#define BM 64
#define BN 256
#define BK 32
#define LDSS 40   // f16 elems per LDS row (32 data + 8 pad; 80B keeps 16B align,
                  // and 20-bank lane stride -> conflict-free b128 reads)

// ---------------------------------------------------------------------------
// Pack x (B,16,64) f32 -> A (B, 1024) f16 with col = n*16 + c  (position-major)
// ---------------------------------------------------------------------------
__global__ __launch_bounds__(256) void pack_x(const float* __restrict__ x,
                                              _Float16* __restrict__ A) {
  int g = blockIdx.x * 256 + threadIdx.x;    // 512*1024 total
  int b = g >> 10;
  int k = g & 1023;
  int n = k >> 4;
  int c = k & 15;
  A[g] = (_Float16)x[b * 1024 + c * 64 + n];
}

// ---------------------------------------------------------------------------
// Materialize one group of W: W[n*64+o][colOff + m*Cg + c] =
//     sign(p[n,m]) * w[o][c][|p[n,m]|-1],  w shape (64, Cg, 64) row-major.
// grid = (64, 64) -> (m, n); block = 256.
// ---------------------------------------------------------------------------
__global__ __launch_bounds__(256) void wmat_fill(const float* __restrict__ w,
                                                 const int* __restrict__ p,
                                                 _Float16* __restrict__ W,
                                                 int Cg, int colOff, int ldw) {
  const int m = blockIdx.x;
  const int n = blockIdx.y;
  const int pe = p[n * 64 + m];
  const int idx = (pe < 0 ? -pe : pe) - 1;
  const float s = pe < 0 ? -1.0f : 1.0f;
  const int tot = 64 * Cg;
  for (int i = threadIdx.x; i < tot; i += 256) {
    int o = i / Cg;
    int c = i - o * Cg;
    float v = s * w[(o * Cg + c) * 64 + idx];
    W[(size_t)(n * 64 + o) * ldw + colOff + m * Cg + c] = (_Float16)v;
  }
}

// ---------------------------------------------------------------------------
// WMMA GEMM: C(512 x 4096) f32 = A(512 x K) f16 * W(4096 x K)^T f16.
// Block = 256 threads (8 waves), tile 64(M) x 256(N); wave tile 32x64
// (2x4 WMMA 16x16x32). Double-buffered LDS; next K-panel staged in VGPRs
// while WMMAs consume the current panel.  grid = (4096/BN, 512/BM) = (16, 8).
// ---------------------------------------------------------------------------
__global__ __launch_bounds__(256) void gemm_f16(const _Float16* __restrict__ A,
                                                const _Float16* __restrict__ W,
                                                float* __restrict__ C, int K) {
  __shared__ __align__(16) _Float16 As[2][BM * LDSS];   // 2 x  5,120 B
  __shared__ __align__(16) _Float16 Ws[2][BN * LDSS];   // 2 x 20,480 B

  const int tid  = threadIdx.x;
  const int wid  = tid >> 5;
  const int lane = tid & 31;
  const int wm   = wid >> 2;        // 0..1  -> M offset wm*32
  const int wn   = wid & 3;         // 0..3  -> N offset wn*64
  const int m0   = blockIdx.y * BM;
  const int n0   = blockIdx.x * BN;
  const int lrow = lane & 15;
  const int half = lane >> 4;

  v8f acc[2][4] = {};

  // global staging coords: A = 1 x b128/thread, W = 4 x b128/thread
  const int arow = tid >> 2;        // 0..63
  const int achk = tid & 3;         // 16B chunk in 64B row
  const _Float16* aptr  = A + (size_t)(m0 + arow) * K + achk * 8;
  const _Float16* wptr  = W + (size_t)(n0 + arow) * K + achk * 8;
  const size_t    wstep = (size_t)64 * K;
  const int ldsr = arow * LDSS + achk * 8;

  uint4 ra, rw0, rw1, rw2, rw3;

  // ---- prologue: stage panel 0 ----
  ra  = *(const uint4*)(aptr);
  rw0 = *(const uint4*)(wptr);
  rw1 = *(const uint4*)(wptr + wstep);
  rw2 = *(const uint4*)(wptr + 2 * wstep);
  rw3 = *(const uint4*)(wptr + 3 * wstep);
  *(uint4*)(&As[0][ldsr]) = ra;
  *(uint4*)(&Ws[0][ldsr]) = rw0;
  *(uint4*)(&Ws[0][ldsr +  64 * LDSS]) = rw1;
  *(uint4*)(&Ws[0][ldsr + 128 * LDSS]) = rw2;
  *(uint4*)(&Ws[0][ldsr + 192 * LDSS]) = rw3;
  __syncthreads();

  const int nk = K / BK;
  for (int kk = 0; kk < nk; ++kk) {
    const int cur = kk & 1;
    const int nxt = cur ^ 1;
    const bool more = (kk + 1) < nk;

    if (more) {  // issue next panel's global loads; latency hidden by WMMAs
      const int ko = (kk + 1) * BK;
      ra  = *(const uint4*)(aptr + ko);
      rw0 = *(const uint4*)(wptr + ko);
      rw1 = *(const uint4*)(wptr + wstep + ko);
      rw2 = *(const uint4*)(wptr + 2 * wstep + ko);
      rw3 = *(const uint4*)(wptr + 3 * wstep + ko);
      if (kk + 2 < nk) {  // pull panel kk+2 toward the WGP caches
        __builtin_prefetch(wptr + ko + BK, 0, 3);
        __builtin_prefetch(wptr + wstep + ko + BK, 0, 3);
        __builtin_prefetch(wptr + 2 * wstep + ko + BK, 0, 3);
        __builtin_prefetch(wptr + 3 * wstep + ko + BK, 0, 3);
      }
    }

    // ---- fragments: lane half h holds K in [8h,8h+8) U [16+8h,24+8h) ----
    v16h afrag[2], wfrag[4];
#pragma unroll
    for (int sm = 0; sm < 2; ++sm) {
      const _Float16* base = &As[cur][(wm * 32 + sm * 16 + lrow) * LDSS];
      Frag16 u;
      u.h[0] = *(const v8h*)(base + half * 8);
      u.h[1] = *(const v8h*)(base + 16 + half * 8);
      afrag[sm] = u.v;
    }
#pragma unroll
    for (int sn = 0; sn < 4; ++sn) {
      const _Float16* base = &Ws[cur][(wn * 64 + sn * 16 + lrow) * LDSS];
      Frag16 u;
      u.h[0] = *(const v8h*)(base + half * 8);
      u.h[1] = *(const v8h*)(base + 16 + half * 8);
      wfrag[sn] = u.v;
    }

#pragma unroll
    for (int sm = 0; sm < 2; ++sm)
#pragma unroll
      for (int sn = 0; sn < 4; ++sn)
        acc[sm][sn] = __builtin_amdgcn_wmma_f32_16x16x32_f16(
            false, afrag[sm], false, wfrag[sn], (short)0, acc[sm][sn],
            false, false);

    if (more) {  // commit staged panel into the other buffer
      *(uint4*)(&As[nxt][ldsr]) = ra;
      *(uint4*)(&Ws[nxt][ldsr]) = rw0;
      *(uint4*)(&Ws[nxt][ldsr +  64 * LDSS]) = rw1;
      *(uint4*)(&Ws[nxt][ldsr + 128 * LDSS]) = rw2;
      *(uint4*)(&Ws[nxt][ldsr + 192 * LDSS]) = rw3;
    }
    __syncthreads();
  }

  // ---- store: C/D layout -> row = 8*half + v, col = lane%16 ----
#pragma unroll
  for (int sm = 0; sm < 2; ++sm)
#pragma unroll
    for (int sn = 0; sn < 4; ++sn) {
      const int col   = n0 + wn * 64 + sn * 16 + lrow;
      const int rbase = m0 + wm * 32 + sm * 16 + half * 8;
#pragma unroll
      for (int v = 0; v < 8; ++v)
        C[(size_t)(rbase + v) * 4096 + col] = acc[sm][sn][v];
    }
}

// ---------------------------------------------------------------------------
// Per-channel (c = j%64) sum / sumsq of |z|/2 over out2 (512x4096).
// grid = 256 blocks, each covers 8192 consecutive floats; i%64 == tid%64.
// ---------------------------------------------------------------------------
__global__ __launch_bounds__(256) void reduce_stats(const float* __restrict__ out2,
                                                    float* __restrict__ stats) {
  __shared__ float ls[256], ls2[256];
  const int tid = threadIdx.x;
  const size_t base = (size_t)blockIdx.x * 8192 + tid;
  float s = 0.f, s2 = 0.f;
#pragma unroll 4
  for (int i = 0; i < 32; ++i) {
    float a = fabsf(out2[base + (size_t)i * 256]) * 0.5f;
    s += a;
    s2 += a * a;
  }
  ls[tid] = s;
  ls2[tid] = s2;
  __syncthreads();
  if (tid < 64) {
    float t  = ls[tid]  + ls[tid + 64]  + ls[tid + 128]  + ls[tid + 192];
    float t2 = ls2[tid] + ls2[tid + 64] + ls2[tid + 128] + ls2[tid + 192];
    atomicAdd(&stats[tid], t);
    atomicAdd(&stats[64 + tid], t2);
  }
}

// ---------------------------------------------------------------------------
// Anew(512 x newK) f16 = [ BN(|out2|/2) (4096 cols) | Aprev (prevK cols) ]
// ---------------------------------------------------------------------------
__global__ __launch_bounds__(256) void bn_apply_concat(
    const float* __restrict__ out2, const float* __restrict__ stats,
    const float* __restrict__ gamma, const float* __restrict__ beta,
    const _Float16* __restrict__ Aprev, int prevK,
    _Float16* __restrict__ Anew, int newK) {
  size_t g = (size_t)blockIdx.x * 256 + threadIdx.x;
  size_t total = (size_t)512 * newK;
  if (g >= total) return;
  int b = (int)(g / newK);
  int j = (int)(g % newK);
  if (j < 4096) {
    int c = j & 63;
    const float inv = 1.0f / 32768.0f;
    float mu  = stats[c] * inv;
    float var = stats[64 + c] * inv - mu * mu;
    float a = fabsf(out2[(size_t)b * 4096 + j]) * 0.5f;
    float h = (a - mu) * rsqrtf(var + 1e-5f) * gamma[c] + beta[c];
    Anew[g] = (_Float16)h;
  } else {
    Anew[g] = Aprev[(size_t)b * prevK + (j - 4096)];
  }
}

// ---------------------------------------------------------------------------
extern "C" void kernel_launch(void* const* d_in, const int* in_sizes, int n_in,
                              void* d_out, int out_size, void* d_ws, size_t ws_size,
                              hipStream_t stream) {
  (void)in_sizes; (void)n_in; (void)out_size; (void)ws_size;

  const float* x     = (const float*)d_in[0];
  const float* w0_0  = (const float*)d_in[1];
  const float* w1_0  = (const float*)d_in[2];
  const float* w1_1  = (const float*)d_in[3];
  const float* w2_0  = (const float*)d_in[4];
  const float* w2_1  = (const float*)d_in[5];
  const float* w2_2  = (const float*)d_in[6];
  const float* bn0_g = (const float*)d_in[7];
  const float* bn0_b = (const float*)d_in[8];
  const float* bn1_g = (const float*)d_in[9];
  const float* bn1_b = (const float*)d_in[10];
  const int*   p0_0  = (const int*)d_in[11];
  const int*   p1_0  = (const int*)d_in[12];
  const int*   p1_1  = (const int*)d_in[13];
  const int*   p2_0  = (const int*)d_in[14];
  const int*   p2_1  = (const int*)d_in[15];
  const int*   p2_2  = (const int*)d_in[16];

  char* ws = (char*)d_ws;
  size_t off = 0;
  _Float16* Wbuf  = (_Float16*)(ws + off); off += (size_t)4096 * 9216 * 2;
  _Float16* Aa    = (_Float16*)(ws + off); off += (size_t)512  * 9216 * 2;
  _Float16* Ab    = (_Float16*)(ws + off); off += (size_t)512  * 9216 * 2;
  float*    O2    = (float*)   (ws + off); off += (size_t)512  * 4096 * 4;
  float*    stats = (float*)   (ws + off);

  const dim3 gW(64, 64);                  // (m, n) pattern cells
  const dim3 gG(4096 / BN, 512 / BM);     // (16, 8)

  // ---- layer 0: K = 1024 ----
  pack_x<<<2048, 256, 0, stream>>>(x, Aa);
  wmat_fill<<<gW, 256, 0, stream>>>(w0_0, p0_0, Wbuf, 16, 0, 1024);
  gemm_f16<<<gG, 256, 0, stream>>>(Aa, Wbuf, O2, 1024);
  hipMemsetAsync(stats, 0, 128 * sizeof(float), stream);
  reduce_stats<<<256, 256, 0, stream>>>(O2, stats);
  bn_apply_concat<<<(512 * 5120) / 256, 256, 0, stream>>>(
      O2, stats, bn0_g, bn0_b, Aa, 1024, Ab, 5120);

  // ---- layer 1: K = 5120 ----
  wmat_fill<<<gW, 256, 0, stream>>>(w1_0, p1_0, Wbuf, 64, 0, 5120);
  wmat_fill<<<gW, 256, 0, stream>>>(w1_1, p1_1, Wbuf, 16, 4096, 5120);
  gemm_f16<<<gG, 256, 0, stream>>>(Ab, Wbuf, O2, 5120);
  hipMemsetAsync(stats, 0, 128 * sizeof(float), stream);
  reduce_stats<<<256, 256, 0, stream>>>(O2, stats);
  bn_apply_concat<<<(512 * 9216) / 256, 256, 0, stream>>>(
      O2, stats, bn1_g, bn1_b, Ab, 5120, Aa, 9216);

  // ---- layer 2: K = 9216, write final f32 output ----
  wmat_fill<<<gW, 256, 0, stream>>>(w2_0, p2_0, Wbuf, 64, 0, 9216);
  wmat_fill<<<gW, 256, 0, stream>>>(w2_1, p2_1, Wbuf, 64, 4096, 9216);
  wmat_fill<<<gW, 256, 0, stream>>>(w2_2, p2_2, Wbuf, 16, 8192, 9216);
  gemm_f16<<<gG, 256, 0, stream>>>(Aa, Wbuf, (float*)d_out, 9216);
}
